// Attention_43301860278871
// MI455X (gfx1250) — matrix operand order repaired and checked by reading, not compiled, not run on hardware
//
#include <hip/hip_runtime.h>

#define B_ 2
#define S_ 2048
#define D_ 2048
#define H_ 16
#define HD_ 128

typedef __attribute__((ext_vector_type(16))) __bf16 v16bf;
typedef __attribute__((ext_vector_type(8)))  float  v8f;
typedef __attribute__((ext_vector_type(4)))  unsigned int u32x4;
typedef __attribute__((ext_vector_type(8)))  unsigned int u32x8;
typedef __attribute__((ext_vector_type(4)))  float  f32x4;

union Frag {
    v16bf v;
    u32x4 q[2];
    unsigned int u[8];
    unsigned short s[16];
};

static __device__ __forceinline__ v8f vzero8() {
    v8f z;
#pragma unroll
    for (int i = 0; i < 8; ++i) z[i] = 0.0f;
    return z;
}

static __device__ __forceinline__ unsigned short f2bf(float x) {
    unsigned int u = __float_as_uint(x);
    return (unsigned short)((u + 0x7FFFu + ((u >> 16) & 1u)) >> 16);  // RNE
}

static __device__ __forceinline__ unsigned int pack2bf(float a, float b) {
    return (unsigned int)f2bf(a) | ((unsigned int)f2bf(b) << 16);
}

// Async global->LDS copy, 16 bytes per lane (GLOBAL_LOAD_ASYNC_TO_LDS_B128,
// GVS addressing: SGPR64 base + VGPR32 offset; LDS dest offset in VGPR).
// Tracked with ASYNCcnt (CDNA5 ISA).
static __device__ __forceinline__ void async_load_b128(unsigned int lds_off,
                                                       unsigned int voff,
                                                       const void* sbase) {
    asm volatile("global_load_async_to_lds_b128 %0, %1, %2"
                 :
                 : "v"(lds_off), "v"(voff), "s"(sbase)
                 : "memory");
}

static __device__ __forceinline__ void wait_async0() {
    asm volatile("s_wait_asynccnt 0x0" ::: "memory");
}

// Generic shared-memory pointer -> 32-bit LDS byte offset (aperture form keeps
// the LDS offset in the low 32 bits).
static __device__ __forceinline__ unsigned int lds_off(const void* p) {
    return (unsigned int)(unsigned long long)p;
}

// Tensor Data Mover: DMA a 2D tile (32 rows x 256 bytes, bf16) from a
// row-major [rows][128] bf16 tensor into LDS with a 16B pad after each 256B
// row (=> 136-element LDS row stride). D# per CDNA5 ISA
// §8.3/§8.4; VADDR2/3 omitted (<=2D). Tracked with TENSORcnt.
static __device__ __forceinline__ void tdm_load_tile_2d(unsigned int lds_base,
                                                        const void* tile_gaddr) {
    const unsigned long long ga = (unsigned long long)tile_gaddr;
    u32x4 g0;
    g0[0] = 1u;                                      // count=1 (valid user D#)
    g0[1] = lds_base;                                // lds_addr (bytes)
    g0[2] = (unsigned int)ga;                        // global_addr[31:0]
    g0[3] = (unsigned int)(ga >> 32) | (2u << 30);   // global_addr[56:32] | type=2
    u32x8 g1;
    g1[0] = (1u << 16)    // data_size = 2 bytes
          | (1u << 20)    // pad_enable
          | (5u << 22)    // pad_interval: 64 DWORDs (one 256B row)
          | (3u << 25);   // pad_amount: 4 DWORDs (16B)
    g1[1] = (unsigned int)HD_ << 16;                 // tensor_dim0 = 128
    g1[2] = 32u << 16;                               // tensor_dim1 = 32
    g1[3] = (unsigned int)HD_ << 16;                 // tile_dim0 = 128
    g1[4] = 32u;                                     // tile_dim1 = 32, tile_dim2 = 0
    g1[5] = (unsigned int)HD_;                       // tensor_dim0_stride = 128
    g1[6] = 0u;
    g1[7] = 0u;
    asm volatile("tensor_load_to_lds %0, %1"
                 :
                 : "s"(g0), "s"(g1)
                 : "memory");
}

static __device__ __forceinline__ void wait_tensor0() {
    __builtin_amdgcn_s_wait_tensorcnt(0);
}

// A-matrix 16x32 bf16 fragment: lane holds M=lane&15 with contiguous
// K = kh*16 .. kh*16+15, kh = lane>=16.
static __device__ __forceinline__ Frag load_frag_a(const unsigned short* base, int ld) {
    const int lane = threadIdx.x & 31;
    const int m = lane & 15;
    const int kh = lane >> 4;
    const unsigned short* p = base + m * ld + kh * 16;
    Frag f;
    f.q[0] = *(const u32x4*)p;
    f.q[1] = *(const u32x4*)(p + 8);
    return f;
}

// B-matrix 32x16 bf16 fragment: lane holds column N=lane&15, contiguous
// K = kh*16 .. kh*16+15 (kh = lane>=16). `base` points at row n=0, stride ld.
static __device__ __forceinline__ Frag load_frag_b(const unsigned short* base, int ld) {
    const int lane = threadIdx.x & 31;
    const int n = lane & 15;
    const int kh = lane >> 4;
    const unsigned short* p = base + n * ld + kh * 16;
    Frag f;
    f.q[0] = *(const u32x4*)p;
    f.q[1] = *(const u32x4*)(p + 8);
    return f;
}

static __device__ __forceinline__ v8f wmma_bf16(const Frag& a, const Frag& b, v8f c) {
    return __builtin_amdgcn_wmma_f32_16x16x32_bf16(false, a.v, false, b.v,
                                                   (short)0, c, false, false);
}

// ---------------------------------------------------------------------------
// Kernel 1: [Q;K;V] = x @ [wqk;wv]^T with fused interleaved RoPE on Q,K.
// Block tile 128(M) x 64(N), 4 waves (2x2), wave tile 64x32, K-step 32.
// Outputs bf16 Q/K/V in [b][h][s][hd] layout for the attention kernel.
// ---------------------------------------------------------------------------
__global__ __launch_bounds__(128) void qkv_rope_kernel(
    const float* __restrict__ x, const float* __restrict__ wqk,
    const float* __restrict__ wv, unsigned short* __restrict__ qb,
    unsigned short* __restrict__ kb, unsigned short* __restrict__ vb) {
    __shared__ __align__(16) unsigned short As[128 * 40];
    __shared__ __align__(16) unsigned short Bs[64 * 40];

    const int t = threadIdx.x;
    const int wid = t >> 5;
    const int m0 = blockIdx.x * 128;
    const int n0 = blockIdx.y * 64;
    const int wm = (wid >> 1) * 64;
    const int wn = (wid & 1) * 32;

    const int nrow = t >> 1, nseg = t & 1;
    const int ng = n0 + nrow;
    const float* wrow = (ng < 2 * D_) ? (wqk + (size_t)ng * D_)
                                      : (wv + (size_t)(ng - 2 * D_) * D_);

    v8f acc[4][2];
#pragma unroll
    for (int mi = 0; mi < 4; ++mi)
#pragma unroll
        for (int ni = 0; ni < 2; ++ni) acc[mi][ni] = vzero8();

    for (int k0 = 0; k0 < D_; k0 += 32) {
        {   // A tile: 128 rows x 32 K, one row per thread, fp32 -> bf16
            const float* srcf = x + (size_t)(m0 + t) * D_ + k0;
            __builtin_prefetch(srcf + 32, 0, 0);  // next K-step
            const f32x4* src = (const f32x4*)srcf;
            unsigned int* dst = (unsigned int*)&As[t * 40];
#pragma unroll
            for (int i = 0; i < 8; ++i) {
                f32x4 f = src[i];
                dst[2 * i] = pack2bf(f.x, f.y);
                dst[2 * i + 1] = pack2bf(f.z, f.w);
            }
        }
        {   // B tile: 64 rows of concatenated [wqk; wv]
            const float* srcf = wrow + k0 + nseg * 16;
            __builtin_prefetch(srcf + 32, 0, 0);
            const f32x4* src = (const f32x4*)srcf;
            unsigned int* dst = (unsigned int*)&Bs[nrow * 40 + nseg * 16];
#pragma unroll
            for (int i = 0; i < 4; ++i) {
                f32x4 f = src[i];
                dst[2 * i] = pack2bf(f.x, f.y);
                dst[2 * i + 1] = pack2bf(f.z, f.w);
            }
        }
        __syncthreads();
#pragma unroll
        for (int mi = 0; mi < 4; ++mi) {
            Frag a = load_frag_a(&As[(wm + mi * 16) * 40], 40);
#pragma unroll
            for (int ni = 0; ni < 2; ++ni) {
                Frag b = load_frag_b(&Bs[(wn + ni * 16) * 40], 40);
                acc[mi][ni] = wmma_bf16(a, b, acc[mi][ni]);
            }
        }
        __syncthreads();
    }

    // Epilogue: RoPE (interleaved even/odd pairs sit in adjacent lanes) + store.
    // inv_freq = 10000^(-2j/HD) = exp2(j * (-2*log2(10000)/HD))
    const float ROPE_C = -0.20762050593046014f;  // -2*log2(10000)/128
    const int lane = t & 31;
    const int col = lane & 15;
    const int rhalf = (lane >> 4) * 8;
#pragma unroll
    for (int ni = 0; ni < 2; ++ni) {
        const int n = n0 + wn + ni * 16 + col;
        const int hd = n & (HD_ - 1);
        const float invf = __builtin_amdgcn_exp2f((float)(hd >> 1) * ROPE_C);
#pragma unroll
        for (int mi = 0; mi < 4; ++mi)
#pragma unroll
            for (int r = 0; r < 8; ++r) {
                const int m = m0 + wm + mi * 16 + rhalf + r;
                float val = acc[mi][ni][r];
                const int bb = m >> 11;          // / S_
                const int spos = m & (S_ - 1);
                float partner = __shfl_xor(val, 1, 32);
                if (n < 2 * D_) {
                    const int h = (n & (D_ - 1)) >> 7;  // / HD_
                    float sn, cs;
                    __sincosf((float)spos * invf, &sn, &cs);
                    float o = (hd & 1) ? (partner * sn + val * cs)
                                       : (val * cs - partner * sn);
                    unsigned short* dst = (n < D_) ? qb : kb;
                    dst[(((size_t)bb * H_ + h) * S_ + spos) * HD_ + hd] = f2bf(o);
                } else {
                    const int c2 = n - 2 * D_;
                    const int h = c2 >> 7;
                    vb[(((size_t)bb * H_ + h) * S_ + spos) * HD_ + (c2 & (HD_ - 1))] =
                        f2bf(val);
                }
            }
    }
}

// ---------------------------------------------------------------------------
// Kernel 2: causal flash attention. Block = (b, h, 64 q rows), 4 waves each
// owning 16 q rows. K/V 32x128 bf16 tiles DMA'd into LDS by the Tensor Data
// Mover (one descriptor pair per iteration, issued by wave 0, TENSORcnt);
// online softmax in fp32 with 16-lane shuffle row reductions; P transposed
// to A-layout via per-wave LDS scratch.
// ---------------------------------------------------------------------------
__global__ __launch_bounds__(128) void flash_attn_kernel(
    const unsigned short* __restrict__ qb, const unsigned short* __restrict__ kb,
    const unsigned short* __restrict__ vb, unsigned short* __restrict__ ab) {
    __shared__ __align__(16) unsigned short Ks[32 * 136];
    __shared__ __align__(16) unsigned short Vs[32 * 136];
    __shared__ __align__(16) unsigned short Ps[4][16 * 40];

    const int t = threadIdx.x;
    const int lane = t & 31;
    const int wid = t >> 5;
    const int qt = blockIdx.x;
    const int h = blockIdx.y;
    const int b = blockIdx.z;
    const int qbase = qt * 64 + wid * 16;
    const int col = lane & 15;
    const int rhalf = (lane >> 4) * 8;
    const float scale = 0.08838834764831845f;  // HD^-0.5
    const size_t headoff = ((size_t)b * H_ + h) * (size_t)S_ * HD_;
    const unsigned short* kbase = kb + headoff;
    const unsigned short* vbase = vb + headoff;
    const unsigned int ks_lds = lds_off(&Ks[0]);
    const unsigned int vs_lds = lds_off(&Vs[0]);

    Frag aq[4];
#pragma unroll
    for (int kk = 0; kk < 4; ++kk)
        aq[kk] = load_frag_a(qb + headoff + (size_t)qbase * HD_ + kk * 32, HD_);

    v8f oacc[8];
#pragma unroll
    for (int ht = 0; ht < 8; ++ht) oacc[ht] = vzero8();
    float mstate[8], lstate[8];
#pragma unroll
    for (int r = 0; r < 8; ++r) { mstate[r] = -3.0e38f; lstate[r] = 0.0f; }

    const int nkt = qt * 2 + 2;  // causal bound: k < (qt+1)*64
    for (int kt = 0; kt < nkt; ++kt) {
        // TDM-stage K/V tile (32 rows x 128 hd, bf16) into padded LDS
        if (wid == 0) {
            tdm_load_tile_2d(ks_lds, kbase + (size_t)kt * 32 * HD_);
            tdm_load_tile_2d(vs_lds, vbase + (size_t)kt * 32 * HD_);
            wait_tensor0();
        }
        __syncthreads();

        // scores: S[16q x 32k] = Q(16x128) . K^T
        v8f sct[2];
#pragma unroll
        for (int nt = 0; nt < 2; ++nt) {
            v8f sc = vzero8();
#pragma unroll
            for (int kk = 0; kk < 4; ++kk) {
                Frag bk = load_frag_b(&Ks[(nt * 16) * 136 + kk * 32], 136);
                sc = wmma_bf16(aq[kk], bk, sc);
            }
            sct[nt] = sc;
        }

        // online softmax over this 32-wide block
#pragma unroll
        for (int r = 0; r < 8; ++r) {
            const int qpos = qbase + rhalf + r;
            const int k0pos = kt * 32 + col;
            float v0 = (k0pos <= qpos) ? sct[0][r] * scale : -3.0e38f;
            float v1 = (k0pos + 16 <= qpos) ? sct[1][r] * scale : -3.0e38f;
            float mx = fmaxf(v0, v1);
#pragma unroll
            for (int off = 1; off < 16; off <<= 1)
                mx = fmaxf(mx, __shfl_xor(mx, off, 32));
            const float nm = fmaxf(mstate[r], mx);
            const float p0 = __expf(v0 - nm);
            const float p1 = __expf(v1 - nm);
            float rs = p0 + p1;
#pragma unroll
            for (int off = 1; off < 16; off <<= 1)
                rs += __shfl_xor(rs, off, 32);
            const float alpha = __expf(mstate[r] - nm);
            lstate[r] = lstate[r] * alpha + rs;
            mstate[r] = nm;
#pragma unroll
            for (int ht = 0; ht < 8; ++ht) oacc[ht][r] *= alpha;
            Ps[wid][(rhalf + r) * 40 + col] = f2bf(p0);
            Ps[wid][(rhalf + r) * 40 + 16 + col] = f2bf(p1);
        }

        // PV: O += P(16x32) . V(32x128)
        Frag ap = load_frag_a(&Ps[wid][0], 40);
#pragma unroll
        for (int ht = 0; ht < 8; ++ht) {
            Frag bv;
#pragma unroll
            for (int j = 0; j < 16; ++j) {
                const int kr = (lane >> 4) * 16 + j;
                bv.s[j] = Vs[kr * 136 + ht * 16 + col];
            }
            oacc[ht] = wmma_bf16(ap, bv, oacc[ht]);
        }
        __syncthreads();
    }

    // normalize and store bf16 attn output in [b][s][h*HD+hd] (= [M][D]) layout
#pragma unroll
    for (int r = 0; r < 8; ++r) {
        const int qpos = qbase + rhalf + r;
        const float inv = 1.0f / lstate[r];
        const size_t rowb = ((size_t)b * S_ + qpos) * D_ + h * HD_;
#pragma unroll
        for (int ht = 0; ht < 8; ++ht)
            ab[rowb + ht * 16 + col] = f2bf(oacc[ht][r] * inv);
    }
}

// ---------------------------------------------------------------------------
// Kernel 3: out = attn(bf16) @ wo^T, fp32 output. A tile staged with async
// global->LDS copies (ASYNCcnt); wo converted fp32->bf16 through VGPRs.
// ---------------------------------------------------------------------------
__global__ __launch_bounds__(128) void out_proj_kernel(
    const unsigned short* __restrict__ ab, const float* __restrict__ wo,
    float* __restrict__ out) {
    __shared__ __align__(16) unsigned short As[128 * 40];
    __shared__ __align__(16) unsigned short Bs[64 * 40];

    const int t = threadIdx.x;
    const int wid = t >> 5;
    const int m0 = blockIdx.x * 128;
    const int n0 = blockIdx.y * 64;
    const int wm = (wid >> 1) * 64;
    const int wn = (wid & 1) * 32;

    const unsigned int adst = lds_off(&As[t * 40]);
    const unsigned int arow = (unsigned int)(((m0 + t) * D_) * 2);

    v8f acc[4][2];
#pragma unroll
    for (int mi = 0; mi < 4; ++mi)
#pragma unroll
        for (int ni = 0; ni < 2; ++ni) acc[mi][ni] = vzero8();

    for (int k0 = 0; k0 < D_; k0 += 32) {
        {   // A tile already bf16: async copy 64B per thread
#pragma unroll
            for (int i = 0; i < 4; ++i)
                async_load_b128(adst + 16 * i, arow + (unsigned int)(k0 * 2) + 16 * i, ab);
        }
        {   // B tile: wo rows fp32 -> bf16
            const int n = t >> 1, seg = t & 1;
            const float* srcf = wo + (size_t)(n0 + n) * D_ + k0 + seg * 16;
            __builtin_prefetch(srcf + 32, 0, 0);
            const f32x4* src = (const f32x4*)srcf;
            unsigned int* dst = (unsigned int*)&Bs[n * 40 + seg * 16];
#pragma unroll
            for (int i = 0; i < 4; ++i) {
                f32x4 f = src[i];
                dst[2 * i] = pack2bf(f.x, f.y);
                dst[2 * i + 1] = pack2bf(f.z, f.w);
            }
        }
        wait_async0();
        __syncthreads();
#pragma unroll
        for (int mi = 0; mi < 4; ++mi) {
            Frag a = load_frag_a(&As[(wm + mi * 16) * 40], 40);
#pragma unroll
            for (int ni = 0; ni < 2; ++ni) {
                Frag b = load_frag_b(&Bs[(wn + ni * 16) * 40], 40);
                acc[mi][ni] = wmma_bf16(a, b, acc[mi][ni]);
            }
        }
        __syncthreads();
    }

    const int lane = t & 31;
    const int col = lane & 15;
    const int rhalf = (lane >> 4) * 8;
#pragma unroll
    for (int mi = 0; mi < 4; ++mi)
#pragma unroll
        for (int ni = 0; ni < 2; ++ni)
#pragma unroll
            for (int r = 0; r < 8; ++r) {
                const int m = m0 + wm + mi * 16 + rhalf + r;
                const int n = n0 + wn + ni * 16 + col;
                out[(size_t)m * D_ + n] = acc[mi][ni][r];
            }
}

extern "C" void kernel_launch(void* const* d_in, const int* in_sizes, int n_in,
                              void* d_out, int out_size, void* d_ws, size_t ws_size,
                              hipStream_t stream) {
    const float* x = (const float*)d_in[0];
    const float* wqk = (const float*)d_in[1];
    const float* wv = (const float*)d_in[2];
    const float* wo = (const float*)d_in[3];

    unsigned short* ws = (unsigned short*)d_ws;
    const size_t per = (size_t)B_ * H_ * S_ * HD_;  // 8.39M bf16 elements
    unsigned short* qb = ws;
    unsigned short* kb = ws + per;
    unsigned short* vb = ws + 2 * per;
    unsigned short* ab = ws + 3 * per;

    dim3 g1(B_ * S_ / 128, 3 * D_ / 64);
    qkv_rope_kernel<<<g1, 128, 0, stream>>>(x, wqk, wv, qb, kb, vb);

    dim3 g2(S_ / 64, H_, B_);
    flash_attn_kernel<<<g2, 128, 0, stream>>>(qb, kb, vb, ab);

    dim3 g3(B_ * S_ / 128, D_ / 64);
    out_proj_kernel<<<g3, 128, 0, stream>>>(ab, wo, (float*)d_out);
}
